// CRFLayer_9414568313310
// MI455X (gfx1250) — matrix operand order repaired and checked by reading, not compiled, not run on hardware
//
#include <hip/hip_runtime.h>
#include <math.h>

#define NUM_TAGS 128
#define BATCH    128
#define SEQ      1024

typedef __attribute__((ext_vector_type(16))) _Float16 v16h;
typedef __attribute__((ext_vector_type(8)))  float    v8f;

#define LOG2E_F  1.4426950408889634f
#define LN2_F    0.6931471805599453f

// Fast transcendental paths: operands are provably normal-range here
// (P's argmax entry is exactly 1.0, so GEMM accumulators are >= e^-5).
__device__ __forceinline__ float fast_exp(float x) {
    return __builtin_amdgcn_exp2f(x * LOG2E_F);   // v_exp_f32
}
__device__ __forceinline__ float fast_log(float x) {
    return __builtin_amdgcn_logf(x) * LN2_F;      // v_log_f32 + v_mul
}

// LDS row strides (in elements). 136 halves = 272B: 16B-aligned fragment loads,
// 68-dword row stride spreads the 64 LDS banks (<=2-way conflict).
#define TSTRIDE 136
#define PSTRIDE 136
#define ASTRIDE 132

struct __align__(16) SmemT {
    _Float16 expT[NUM_TAGS * TSTRIDE];   // exp(transitions), f16, 34,816 B
    _Float16 P[16 * PSTRIDE];            // exp(alpha - rowmax), f16,  4,352 B
    float    alpha[16 * ASTRIDE];        // log-domain alpha,          8,448 B
    float    redMax[16 * 16];            // rowmax partials
    float    redSum[16 * 16];            // sumexp partials
    float    rowMaxFull[16];             // reduced rowmax per batch row
};                                       // ~49.7 KB of 320 KB WGP LDS

union V16U { v16h v; int4 q[2]; };

// ---------------------------------------------------------------------------
// Partition function: one workgroup handles 16 batch rows for the full scan.
// Per step: rescale -> f16 GEMM via V_WMMA_F32_16X16X32_F16 -> log + emission.
// ---------------------------------------------------------------------------
__global__ __launch_bounds__(256) void crf_partition_kernel(
    const float* __restrict__ emissions,         // [BATCH, SEQ, NUM_TAGS]
    const unsigned char* __restrict__ mask,      // [BATCH, SEQ]
    const float* __restrict__ transitions,       // [NUM_TAGS, NUM_TAGS]
    const float* __restrict__ start_t,           // [NUM_TAGS]
    const float* __restrict__ end_t,             // [NUM_TAGS]
    float* __restrict__ logZ)                    // [BATCH]
{
    __shared__ SmemT sm;
    const int tid   = threadIdx.x;
    const int lane  = tid & 31;
    const int wave  = tid >> 5;          // 0..7 : owns j-tile [16*wave, 16*wave+16)
    const int bbase = blockIdx.x * 16;   // 8 blocks x 16 batch rows

    // --- expT = exp(T) in f16 into LDS (loop-invariant B matrix) ---
    for (int idx = tid; idx < NUM_TAGS * NUM_TAGS; idx += 256) {
        int i = idx >> 7, j = idx & 127;
        sm.expT[i * TSTRIDE + j] = (_Float16)fast_exp(transitions[idx]);
    }
    // --- alpha0 = start_t + emissions[:,0,:] ---
    {
        int row = tid >> 4, seg = tid & 15;
        const float* e0 = emissions + (size_t)(bbase + row) * SEQ * NUM_TAGS;
        for (int q = 0; q < 8; ++q) {
            int j = seg * 8 + q;
            sm.alpha[row * ASTRIDE + j] = start_t[j] + e0[j];
        }
    }
    __syncthreads();

    // --- B fragments: expT K-chunks for this wave's j-tile, resident in VGPRs.
    //     B(KxN) layout: lane&15 = N; halves 0-7: K = 8*(lane>>4)+0..7, halves 8-15: K = 16+8*(lane>>4)+0..7.
    v16h bfrag[4];
    #pragma unroll
    for (int kc = 0; kc < 4; ++kc) {
        const int kb = kc * 32 + (lane >> 4) * 8;
        const int ncol = wave * 16 + (lane & 15);
        V16U u;
        #pragma unroll
        for (int s = 0; s < 16; ++s)
            u.v[s] = sm.expT[(kb + (s & 7) + (s >> 3) * 16) * TSTRIDE + ncol];
        bfrag[kc] = u.v;
    }

    const int row16 = tid >> 4, seg = tid & 15;     // rowmax/P work split
    const int n = lane & 15, msel = lane >> 4;      // WMMA C/D layout coords

    for (int t = 1; t < SEQ; ++t) {
        __syncthreads();   // alpha writes of step t-1 & red arrays reuse

        // (a) partial row max of alpha
        float pm = -INFINITY;
        for (int q = 0; q < 8; ++q)
            pm = fmaxf(pm, sm.alpha[row16 * ASTRIDE + seg * 8 + q]);
        sm.redMax[row16 * 16 + seg] = pm;
        __syncthreads();

        // (b) reduce rowmax, emit P = exp(alpha - rowmax) in f16
        float rmax = sm.redMax[row16 * 16];
        for (int k = 1; k < 16; ++k) rmax = fmaxf(rmax, sm.redMax[row16 * 16 + k]);
        if (seg == 0) sm.rowMaxFull[row16] = rmax;
        for (int q = 0; q < 8; ++q) {
            int j = seg * 8 + q;
            sm.P[row16 * PSTRIDE + j] =
                (_Float16)fast_exp(sm.alpha[row16 * ASTRIDE + j] - rmax);
        }
        __syncthreads();

        // (c) C(16x16) = P(16x128) x expT(128x16), K-loop of 4 WMMAs.
        //     A(16x32) layout: lane&15 = M; halves 0-7: K = 8*msel+0..7,
        //     halves 8-15: K = 16 + 8*msel + 0..7.
        v8f c = {};
        #pragma unroll
        for (int kc = 0; kc < 4; ++kc) {
            const int koff = kc * 32 + msel * 8;
            V16U ua;
            ua.q[0] = *(const int4*)&sm.P[(lane & 15) * PSTRIDE + koff];
            ua.q[1] = *(const int4*)&sm.P[(lane & 15) * PSTRIDE + koff + 16];
            c = __builtin_amdgcn_wmma_f32_16x16x32_f16(
                    false, ua.v, false, bfrag[kc], (short)0, c, false, false);
        }

        // (d) alpha' = log(c) + rowmax + emission, gated by mask.
        //     C/D layout: VGPR r -> M = r + 8*msel, N = lane&15.
        #pragma unroll
        for (int r = 0; r < 8; ++r) {
            const int m = r + 8 * msel;
            const int j = wave * 16 + n;
            const int b = bbase + m;
            const size_t eidx = ((size_t)b * SEQ + t) * NUM_TAGS + j;
            const float e  = emissions[eidx];
            const bool mk  = mask[(size_t)b * SEQ + t] != 0;
            const float od = sm.alpha[m * ASTRIDE + j];
            const float nw = fast_log(c[r]) + sm.rowMaxFull[m] + e;
            sm.alpha[m * ASTRIDE + j] = mk ? nw : od;
            if (t + 1 < SEQ)   // pull next step's emission line into WGP cache
                __builtin_prefetch(&emissions[eidx + NUM_TAGS], 0, 3);
        }
    }

    // --- logZ[b] = logsumexp_j(alpha[b,j] + end_t[j]) ---
    __syncthreads();
    float pm = -INFINITY;
    for (int q = 0; q < 8; ++q) {
        int j = seg * 8 + q;
        pm = fmaxf(pm, sm.alpha[row16 * ASTRIDE + j] + end_t[j]);
    }
    sm.redMax[row16 * 16 + seg] = pm;
    __syncthreads();
    float rmax = sm.redMax[row16 * 16];
    for (int k = 1; k < 16; ++k) rmax = fmaxf(rmax, sm.redMax[row16 * 16 + k]);
    float ps = 0.f;
    for (int q = 0; q < 8; ++q) {
        int j = seg * 8 + q;
        ps += fast_exp(sm.alpha[row16 * ASTRIDE + j] + end_t[j] - rmax);
    }
    sm.redSum[row16 * 16 + seg] = ps;
    __syncthreads();
    if (seg == 0) {
        float s = 0.f;
        for (int k = 0; k < 16; ++k) s += sm.redSum[row16 * 16 + k];
        logZ[bbase + row16] = fast_log(s) + rmax;
    }
}

// ---------------------------------------------------------------------------
// Gold path score: one wave32 per batch row, lanes stride the sequence.
// ---------------------------------------------------------------------------
__global__ __launch_bounds__(32) void crf_gold_kernel(
    const float* __restrict__ emissions,
    const int* __restrict__ tags,
    const unsigned char* __restrict__ mask,
    const float* __restrict__ transitions,
    const float* __restrict__ start_t,
    const float* __restrict__ end_t,
    float* __restrict__ gold)
{
    const int b = blockIdx.x;
    const int l = threadIdx.x;
    const float* em = emissions + (size_t)b * SEQ * NUM_TAGS;
    const int* tg = tags + (size_t)b * SEQ;
    const unsigned char* mk = mask + (size_t)b * SEQ;

    float part = 0.f;
    int cnt = 0;
    for (int t = l; t < SEQ; t += 32) cnt += mk[t] ? 1 : 0;
    for (int t = 1 + l; t < SEQ; t += 32) {
        if (mk[t]) {
            int tp = tg[t - 1], tc = tg[t];
            part += transitions[tp * NUM_TAGS + tc] + em[(size_t)t * NUM_TAGS + tc];
        }
    }
    #pragma unroll
    for (int off = 16; off > 0; off >>= 1) {
        part += __shfl_down(part, off);
        cnt  += __shfl_down(cnt, off);
    }
    if (l == 0) {
        const int t0 = tg[0];
        float s = start_t[t0] + em[t0] + part;
        const int last = cnt - 1;
        s += end_t[tg[last]];
        gold[b] = s;
    }
}

// ---------------------------------------------------------------------------
// Final: mean(logZ - gold)
// ---------------------------------------------------------------------------
__global__ __launch_bounds__(128) void crf_final_kernel(
    const float* __restrict__ logZ, const float* __restrict__ gold,
    float* __restrict__ out)
{
    __shared__ float red[BATCH];
    const int t = threadIdx.x;
    red[t] = logZ[t] - gold[t];
    __syncthreads();
    for (int s = 64; s > 0; s >>= 1) {
        if (t < s) red[t] += red[t + s];
        __syncthreads();
    }
    if (t == 0) out[0] = red[0] / (float)BATCH;
}

extern "C" void kernel_launch(void* const* d_in, const int* in_sizes, int n_in,
                              void* d_out, int out_size, void* d_ws, size_t ws_size,
                              hipStream_t stream) {
    const float* emissions         = (const float*)d_in[0];
    const int* tags                = (const int*)d_in[1];
    const unsigned char* mask      = (const unsigned char*)d_in[2];
    const float* transitions       = (const float*)d_in[3];
    const float* start_transitions = (const float*)d_in[4];
    const float* end_transitions   = (const float*)d_in[5];
    float* out  = (float*)d_out;
    float* logZ = (float*)d_ws;          // [BATCH]
    float* gold = logZ + BATCH;          // [BATCH]

    crf_partition_kernel<<<BATCH / 16, 256, 0, stream>>>(
        emissions, mask, transitions, start_transitions, end_transitions, logZ);
    crf_gold_kernel<<<BATCH, 32, 0, stream>>>(
        emissions, tags, mask, transitions, start_transitions, end_transitions, gold);
    crf_final_kernel<<<1, 128, 0, stream>>>(logZ, gold, out);
}
